// ResidualAttentionBlock_18769007084336
// MI455X (gfx1250) — matrix operand order repaired and checked by reading, not compiled, not run on hardware
//
#include <hip/hip_runtime.h>
#include <hip/hip_bf16.h>

#define BB 2
#define NN 2048
#define WW 1024
#define HH 16
#define HDIM 64
#define MM (BB*NN)        // 4096 rows
#define QKV3 (3*WW)       // 3072

typedef __attribute__((ext_vector_type(16))) __bf16 v16bf;
typedef __attribute__((ext_vector_type(8)))  float  v8f;

union FragU { uint4 u[2]; v16bf v; };

__device__ inline v8f v8zero() {
  v8f z = {0.f,0.f,0.f,0.f,0.f,0.f,0.f,0.f};
  return z;
}

__device__ inline v8f wmma_bf16(v16bf a, v16bf b, v8f c) {
  // (neg_a, A, neg_b, B, c_mod, C, reuse_a, reuse_b)
  return __builtin_amdgcn_wmma_f32_16x16x32_bf16(false, a, false, b, (short)0, c, false, false);
}

// ---------- DPP 16-lane row reductions (pure VALU, no LDS traffic) --------
// quad_perm xor1 = 0xB1, xor2 = 0x4E, row_half_mirror = 0x141, row_mirror = 0x140
__device__ inline float red_max16(float x) {
  int t;
  t = __builtin_amdgcn_update_dpp(0, __float_as_int(x), 0xB1,  0xF, 0xF, true);
  x = fmaxf(x, __int_as_float(t));
  t = __builtin_amdgcn_update_dpp(0, __float_as_int(x), 0x4E,  0xF, 0xF, true);
  x = fmaxf(x, __int_as_float(t));
  t = __builtin_amdgcn_update_dpp(0, __float_as_int(x), 0x141, 0xF, 0xF, true);
  x = fmaxf(x, __int_as_float(t));
  t = __builtin_amdgcn_update_dpp(0, __float_as_int(x), 0x140, 0xF, 0xF, true);
  x = fmaxf(x, __int_as_float(t));
  return x;
}
__device__ inline float red_sum16(float x) {
  int t;
  t = __builtin_amdgcn_update_dpp(0, __float_as_int(x), 0xB1,  0xF, 0xF, true);
  x += __int_as_float(t);
  t = __builtin_amdgcn_update_dpp(0, __float_as_int(x), 0x4E,  0xF, 0xF, true);
  x += __int_as_float(t);
  t = __builtin_amdgcn_update_dpp(0, __float_as_int(x), 0x141, 0xF, 0xF, true);
  x += __int_as_float(t);
  t = __builtin_amdgcn_update_dpp(0, __float_as_int(x), 0x140, 0xF, 0xF, true);
  x += __int_as_float(t);
  return x;
}

// ---------- async global->LDS copy (16 bytes / lane), ASYNCcnt tracked ----
__device__ inline void async_cp16(unsigned lds_off, const void* gptr) {
  asm volatile("global_load_async_to_lds_b128 %0, %1, off"
               :: "v"(lds_off), "v"(gptr) : "memory");
}
__device__ inline void wait_async0() {
  asm volatile("s_wait_asynccnt 0" ::: "memory");
}

// ---- A fragment: 16x32 bf16, row-major source (global or LDS).
// lane: row = l&15, k = (v>>2)*16 + (l>>4)*8 + (v&3)*2 -> two 16B loads.
__device__ inline v16bf load_a_frag(const __bf16* A, int lda,
                                    int row, int kbase, int lane) {
  FragU f;
  const __bf16* p = A + (size_t)row * lda + kbase + ((lane >> 4) << 3);
  f.u[0] = *(const uint4*)(p);
  f.u[1] = *(const uint4*)(p + 16);
  return f.v;
}

// ---- B fragment from pre-packed tiles: tile = 512 bf16, [group][lane][8]
__device__ inline v16bf load_b_packed(const __bf16* Bpk, long tileIdx, int lane) {
  FragU f;
  const uint4* p = (const uint4*)Bpk + tileIdx * 64;
  f.u[0] = p[lane];
  f.u[1] = p[32 + lane];
  return f.v;
}

// ---- B fragment where B columns are rows of a row-major matrix (K^T)
__device__ inline v16bf load_b_rows_frag(const __bf16* basep, long rstride,
                                         int col_base, int koff, int lane) {
  FragU f;
  const __bf16* p = basep + (long)(col_base + (lane & 15)) * rstride
                  + koff + ((lane >> 4) << 3);
  f.u[0] = *(const uint4*)(p);
  f.u[1] = *(const uint4*)(p + 16);
  return f.v;
}

// ================= weight pack: f32 [K,N] -> bf16 B-fragments =============
__global__ __launch_bounds__(256)
void pack_b_kernel(const float* __restrict__ w, __bf16* __restrict__ dst,
                   int Kd, int Nd) {
  int tid = blockIdx.x * 256 + threadIdx.x;
  int total = (Kd >> 5) * (Nd >> 4) * 64;
  if (tid >= total) return;
  int lane = tid & 31;
  int g    = (tid >> 5) & 1;
  int t    = tid >> 6;
  int ntiles = Nd >> 4;
  int nt = t % ntiles, kt = t / ntiles;
  int col = nt * 16 + (lane & 15);
  int kb  = kt * 32 + ((lane >> 4) << 3) + g * 16;
  __bf16* o = dst + (long)t * 512 + g * 256 + lane * 8;
#pragma unroll
  for (int j = 0; j < 8; ++j)
    o[j] = (__bf16)w[(long)(kb + j) * Nd + col];
}

// ====== V pack: bf16 qkv -> per-(b,h) B-fragment tiles (s-dim = k-dim) ====
__global__ __launch_bounds__(256)
void pack_v_kernel(const __bf16* __restrict__ qkv, __bf16* __restrict__ vpk) {
  int tid = blockIdx.x * 256 + threadIdx.x;
  int lane = tid & 31;
  int g    = (tid >> 5) & 1;
  int gt   = tid >> 6;          // global tile 0..8191
  int t    = gt & 255;          // tile within (b,h): kt*4 + nt
  int bh   = gt >> 8;           // 0..31
  int b = bh >> 4, h = bh & 15;
  int kt = t >> 2, nt = t & 3;
  int col = nt * 16 + (lane & 15);
  int kb  = kt * 32 + ((lane >> 4) << 3) + g * 16;
  const __bf16* src = qkv + (long)b * NN * QKV3 + h * (3 * HDIM) + 2 * HDIM;
  __bf16* o = vpk + (long)gt * 512 + g * 256 + lane * 8;
#pragma unroll
  for (int j = 0; j < 8; ++j)
    o[j] = src[(long)(kb + j) * QKV3 + col];
}

// ================= LayerNorm + cast to bf16 ===============================
__global__ __launch_bounds__(256)
void ln_cast_kernel(const float* __restrict__ x, const float* __restrict__ g,
                    const float* __restrict__ b, __bf16* __restrict__ out) {
  __shared__ float ws[8], wss[8];
  int row = blockIdx.x;
  const float* xr = x + (long)row * WW;
  float s = 0.f, ss = 0.f;
  for (int i = threadIdx.x; i < WW; i += 256) {
    float v = xr[i]; s += v; ss += v * v;
  }
#pragma unroll
  for (int m = 16; m >= 1; m >>= 1) {
    s  += __shfl_xor(s,  m, 32);
    ss += __shfl_xor(ss, m, 32);
  }
  if ((threadIdx.x & 31) == 0) { ws[threadIdx.x >> 5] = s; wss[threadIdx.x >> 5] = ss; }
  __syncthreads();
  float ts = 0.f, tss = 0.f;
#pragma unroll
  for (int i = 0; i < 8; ++i) { ts += ws[i]; tss += wss[i]; }
  float mean = ts * (1.f / WW);
  float var  = tss * (1.f / WW) - mean * mean;
  float rstd = rsqrtf(var + 1e-5f);
  for (int i = threadIdx.x; i < WW; i += 256) {
    float v = (xr[i] - mean) * rstd * g[i] + b[i];
    out[(long)row * WW + i] = (__bf16)v;
  }
}

// ====== WMMA GEMM: 128x128 block tile, 32x64 wave tile (8 WMMA/k-step), ===
// ====== async double-buffered LDS staging =================================
// EPI: 0 = bias -> bf16 ; 1 = bias+GELU -> bf16 ; 2 = bias+residual -> f32
template <int EPI>
__global__ __launch_bounds__(256)
void gemm_kernel(const __bf16* __restrict__ A, const __bf16* __restrict__ Bpk,
                 const float* __restrict__ bias, const float* __restrict__ res,
                 __bf16* __restrict__ outb, float* __restrict__ outf,
                 int Nd, int Kd) {
  __shared__ __align__(16) __bf16 sA[2][128 * 32];   // 8KB per buffer
  __shared__ __align__(16) __bf16 sB[2][8 * 512];    // 8KB per buffer

  const int tid  = threadIdx.x;
  const int lane = tid & 31;
  const int wave = tid >> 5;
  const int wm = wave >> 1, wn = wave & 1;    // 4 x 2 wave grid
  const int mrow0  = blockIdx.x * 128;
  const int ntile0 = blockIdx.y * 8;          // 128-wide block = 8 n-tiles
  const int ntiles_n = Nd >> 4;

  const unsigned ldsA[2] = {(unsigned)(size_t)&sA[0][0], (unsigned)(size_t)&sA[1][0]};
  const unsigned ldsB[2] = {(unsigned)(size_t)&sB[0][0], (unsigned)(size_t)&sB[1][0]};

  auto stage = [&](int kt, int bi) {
    // A slab: 128 rows x 32 cols bf16 = 512 x 16B chunks, 2 per thread
#pragma unroll
    for (int i = 0; i < 2; ++i) {
      int c = tid + i * 256;
      int r = c >> 2, part = c & 3;
      const __bf16* g = A + (long)(mrow0 + r) * Kd + kt * 32 + part * 8;
      async_cp16(ldsA[bi] + r * 64 + part * 16, g);
    }
    // B slab: 8 packed tiles x 1KB = 512 x 16B chunks, 2 per thread
#pragma unroll
    for (int i = 0; i < 2; ++i) {
      int c = tid + i * 256;
      int j = c >> 6, w = c & 63;
      const __bf16* g = Bpk + ((long)kt * ntiles_n + ntile0 + j) * 512 + w * 8;
      async_cp16(ldsB[bi] + j * 1024 + w * 16, g);
    }
  };

  v8f acc[2][4];
#pragma unroll
  for (int mi = 0; mi < 2; ++mi)
#pragma unroll
    for (int ni = 0; ni < 4; ++ni) acc[mi][ni] = v8zero();

  const int kt_end = Kd >> 5;
  stage(0, 0);
  for (int kt = 0; kt < kt_end; ++kt) {
    wait_async0();
    __syncthreads();
    if (kt + 1 < kt_end) stage(kt + 1, (kt + 1) & 1);

    const __bf16* cA = sA[kt & 1];
    const __bf16* cB = sB[kt & 1];
    v16bf a0 = load_a_frag(cA, 32, wm * 32 +      (lane & 15), 0, lane);
    v16bf a1 = load_a_frag(cA, 32, wm * 32 + 16 + (lane & 15), 0, lane);
    FragU b[4];
#pragma unroll
    for (int ni = 0; ni < 4; ++ni) {
      const uint4* p = (const uint4*)(cB + (wn * 4 + ni) * 512);
      b[ni].u[0] = p[lane];
      b[ni].u[1] = p[32 + lane];
    }
#pragma unroll
    for (int ni = 0; ni < 4; ++ni) {
      acc[0][ni] = wmma_bf16(a0, b[ni].v, acc[0][ni]);
      acc[1][ni] = wmma_bf16(a1, b[ni].v, acc[1][ni]);
    }
  }

#pragma unroll
  for (int mi = 0; mi < 2; ++mi) {
#pragma unroll
    for (int ni = 0; ni < 4; ++ni) {
      int col = ntile0 * 16 + wn * 64 + ni * 16 + (lane & 15);
      float bv = bias[col];
#pragma unroll
      for (int v = 0; v < 8; ++v) {
        int row = mrow0 + wm * 32 + mi * 16 + ((lane >> 4) << 3) + v;
        float x = acc[mi][ni][v] + bv;
        long off = (long)row * Nd + col;
        if (EPI == 1) x = 0.5f * x * (1.f + erff(x * 0.70710678f));
        if (EPI == 2) outf[off] = res[off] + x;
        else          outb[off] = (__bf16)x;
      }
    }
  }
}

// ================= Flash attention: 1 wave per (b, h, 16-row q-tile) ======
__global__ __launch_bounds__(256)
void attn_kernel(const __bf16* __restrict__ qkv, const __bf16* __restrict__ vpk,
                 __bf16* __restrict__ outb) {
  __shared__ __align__(16) __bf16 pstage[8][16 * 32];
  const int lane = threadIdx.x & 31;
  const int wave = threadIdx.x >> 5;
  const int tt = blockIdx.x * 8 + wave;   // 0..127 query tiles
  const int h  = blockIdx.y;              // head
  const int b  = blockIdx.z;              // batch
  const long rstride = QKV3;
  const __bf16* base  = qkv + (long)b * NN * rstride + h * (3 * HDIM);
  const __bf16* kbase = base + HDIM;
  const __bf16* vbpk  = vpk + (long)(b * HH + h) * 256 * 512;

  // Q fragments (head-dim 64 = two 32-wide k-steps)
  v16bf qf0, qf1;
  {
    const __bf16* qp = base + (long)(tt * 16 + (lane & 15)) * rstride;
    FragU f;
    const __bf16* p0 = qp + ((lane >> 4) << 3);
    f.u[0] = *(const uint4*)(p0);      f.u[1] = *(const uint4*)(p0 + 16);
    qf0 = f.v;
    const __bf16* p1 = qp + 32 + ((lane >> 4) << 3);
    f.u[0] = *(const uint4*)(p1);      f.u[1] = *(const uint4*)(p1 + 16);
    qf1 = f.v;
  }

  v8f o0 = v8zero(), o1 = v8zero(), o2 = v8zero(), o3 = v8zero();
  float m[8], ssum[8];
#pragma unroll
  for (int v = 0; v < 8; ++v) { m[v] = -3.0e38f; ssum[v] = 0.f; }

  for (int s0 = 0; s0 < NN; s0 += 32) {
    // S = (q*k)/8 over this 16x32 slab of keys
    v8f st0 = v8zero(), st1 = v8zero();
    st0 = wmma_bf16(qf0, load_b_rows_frag(kbase, rstride, s0,      0,  lane), st0);
    st0 = wmma_bf16(qf1, load_b_rows_frag(kbase, rstride, s0,      32, lane), st0);
    st1 = wmma_bf16(qf0, load_b_rows_frag(kbase, rstride, s0 + 16, 0,  lane), st1);
    st1 = wmma_bf16(qf1, load_b_rows_frag(kbase, rstride, s0 + 16, 32, lane), st1);

    float alpha[8];
#pragma unroll
    for (int v = 0; v < 8; ++v) {
      float a = st0[v] * 0.125f, c = st1[v] * 0.125f;
      st0[v] = a; st1[v] = c;
      float mx = red_max16(fmaxf(a, c));          // DPP row reduction
      float mn = fmaxf(m[v], mx);
      alpha[v] = __expf(m[v] - mn);
      m[v] = mn;
    }
    // P = exp(S - m): stage to LDS (C-layout -> A-layout), accumulate rowsum
#pragma unroll
    for (int v = 0; v < 8; ++v) {
      float p0 = __expf(st0[v] - m[v]);
      float p1 = __expf(st1[v] - m[v]);
      int row = v + ((lane >> 4) << 3);
      pstage[wave][row * 32 + (lane & 15)]      = (__bf16)p0;
      pstage[wave][row * 32 + 16 + (lane & 15)] = (__bf16)p1;
      ssum[v] = ssum[v] * alpha[v] + red_sum16(p0 + p1);
    }
    asm volatile("s_wait_dscnt 0" ::: "memory");   // RAW: LDS stores -> loads

    FragU pf;
    {
      const __bf16* p = &pstage[wave][(lane & 15) * 32 + ((lane >> 4) << 3)];
      pf.u[0] = *(const uint4*)(p);
      pf.u[1] = *(const uint4*)(p + 16);
    }
    asm volatile("s_wait_dscnt 0" ::: "memory");   // loads done before next overwrite

    // O = O*alpha + P @ V   (V from pre-packed fragments: 2x b128 each)
#pragma unroll
    for (int v = 0; v < 8; ++v) {
      o0[v] *= alpha[v]; o1[v] *= alpha[v]; o2[v] *= alpha[v]; o3[v] *= alpha[v];
    }
    int ktile = (s0 >> 5) * 4;
    o0 = wmma_bf16(pf.v, load_b_packed(vbpk, ktile + 0, lane), o0);
    o1 = wmma_bf16(pf.v, load_b_packed(vbpk, ktile + 1, lane), o1);
    o2 = wmma_bf16(pf.v, load_b_packed(vbpk, ktile + 2, lane), o2);
    o3 = wmma_bf16(pf.v, load_b_packed(vbpk, ktile + 3, lane), o3);
  }

  // normalize and store bf16 into [row, h*64 + hd] layout
  float inv[8];
#pragma unroll
  for (int v = 0; v < 8; ++v) inv[v] = 1.f / ssum[v];
  v8f os[4] = {o0, o1, o2, o3};
#pragma unroll
  for (int ht = 0; ht < 4; ++ht) {
#pragma unroll
    for (int v = 0; v < 8; ++v) {
      int row = b * NN + tt * 16 + v + ((lane >> 4) << 3);
      int col = h * HDIM + ht * 16 + (lane & 15);
      outb[(long)row * WW + col] = (__bf16)(os[ht][v] * inv[v]);
    }
  }
}

// ========================= launcher =======================================
extern "C" void kernel_launch(void* const* d_in, const int* in_sizes, int n_in,
                              void* d_out, int out_size, void* d_ws, size_t ws_size,
                              hipStream_t stream) {
  const float* x      = (const float*)d_in[0];
  const float* ln1_g  = (const float*)d_in[1];
  const float* ln1_b  = (const float*)d_in[2];
  const float* qkv_w  = (const float*)d_in[3];
  const float* qkv_b  = (const float*)d_in[4];
  const float* proj_w = (const float*)d_in[5];
  const float* proj_b = (const float*)d_in[6];
  const float* ln2_g  = (const float*)d_in[7];
  const float* ln2_b  = (const float*)d_in[8];
  const float* fc_w   = (const float*)d_in[9];
  const float* fc_b   = (const float*)d_in[10];
  const float* fc2_w  = (const float*)d_in[11];
  const float* fc2_b  = (const float*)d_in[12];
  float* out = (float*)d_out;

  char* ws = (char*)d_ws;
  size_t off = 0;
  auto alloc = [&](size_t bytes) { char* p = ws + off; off += (bytes + 255) & ~size_t(255); return p; };
  __bf16* qkvw_pk = (__bf16*)alloc((size_t)WW * QKV3 * 2);
  __bf16* projw_pk= (__bf16*)alloc((size_t)WW * WW * 2);
  __bf16* fcw_pk  = (__bf16*)alloc((size_t)WW * 4 * WW * 2);
  __bf16* fc2w_pk = (__bf16*)alloc((size_t)4 * WW * WW * 2);
  __bf16* ln1x    = (__bf16*)alloc((size_t)MM * WW * 2);
  __bf16* qkv     = (__bf16*)alloc((size_t)MM * QKV3 * 2);
  __bf16* vpk     = (__bf16*)alloc((size_t)BB * HH * NN * HDIM * 2);
  __bf16* attn    = (__bf16*)alloc((size_t)MM * WW * 2);
  float*  x1      = (float*) alloc((size_t)MM * WW * 4);
  __bf16* ln2x    = (__bf16*)alloc((size_t)MM * WW * 2);
  __bf16* hbuf    = (__bf16*)alloc((size_t)MM * 4 * WW * 2);
  (void)ws_size; (void)in_sizes; (void)n_in; (void)out_size;

  // pack weights -> bf16 B fragments
  pack_b_kernel<<<((WW/32)*(QKV3/16)*64 + 255)/256, 256, 0, stream>>>(qkv_w,  qkvw_pk, WW, QKV3);
  pack_b_kernel<<<((WW/32)*(WW/16)*64   + 255)/256, 256, 0, stream>>>(proj_w, projw_pk, WW, WW);
  pack_b_kernel<<<((WW/32)*(4*WW/16)*64 + 255)/256, 256, 0, stream>>>(fc_w,   fcw_pk, WW, 4*WW);
  pack_b_kernel<<<((4*WW/32)*(WW/16)*64 + 255)/256, 256, 0, stream>>>(fc2_w,  fc2w_pk, 4*WW, WW);

  // ln1(x) -> bf16
  ln_cast_kernel<<<MM, 256, 0, stream>>>(x, ln1_g, ln1_b, ln1x);

  // qkv = ln1x @ qkv_w + b  (bf16 out)
  gemm_kernel<0><<<dim3(MM/128, QKV3/128), 256, 0, stream>>>(
      ln1x, qkvw_pk, qkv_b, nullptr, qkv, nullptr, QKV3, WW);

  // pack V into per-(b,h) WMMA B-fragments
  pack_v_kernel<<<(BB*HH*256*64)/256, 256, 0, stream>>>(qkv, vpk);

  // attention
  attn_kernel<<<dim3(NN/16/8, HH, BB), 256, 0, stream>>>(qkv, vpk, attn);

  // x1 = x + attn @ proj_w + b  (f32 out)
  gemm_kernel<2><<<dim3(MM/128, WW/128), 256, 0, stream>>>(
      attn, projw_pk, proj_b, x, nullptr, x1, WW, WW);

  // ln2(x1) -> bf16
  ln_cast_kernel<<<MM, 256, 0, stream>>>(x1, ln2_g, ln2_b, ln2x);

  // h = gelu(ln2x @ fc_w + b)  (bf16 out)
  gemm_kernel<1><<<dim3(MM/128, 4*WW/128), 256, 0, stream>>>(
      ln2x, fcw_pk, fc_b, nullptr, hbuf, nullptr, 4*WW, WW);

  // out = x1 + h @ fc2_w + b  (f32 out)
  gemm_kernel<2><<<dim3(MM/128, WW/128), 256, 0, stream>>>(
      hbuf, fc2w_pk, fc2_b, x1, nullptr, out, WW, 4*WW);
}
